// TransformerClassifier_21792664060295
// MI455X (gfx1250) — matrix operand order repaired and checked by reading, not compiled, not run on hardware
//
#include <hip/hip_runtime.h>

typedef __attribute__((ext_vector_type(16))) _Float16 v16h;
typedef __attribute__((ext_vector_type(8)))  _Float16 v8h;
typedef __attribute__((ext_vector_type(4)))  _Float16 v4h;
typedef __attribute__((ext_vector_type(8)))  float    v8f;
typedef __attribute__((ext_vector_type(4)))  int      v4i;

#define TC_B   8
#define TC_L   512
#define TC_DIN 256
#define TC_E   512
#define TC_H   8
#define TC_FF  2048
#define TC_C   10
#define TC_NL  4
#define TC_HD  64
#define TC_M   (TC_B * TC_L)   // 4096 tokens

#define BM  128
#define BN  64
#define BK  32
#define LDP 40   // padded LDS row pitch in halves (80B -> conflict-free, 16B aligned)

// ---------------------------------------------------------------------------
// Async global->LDS staging (CDNA5 path, ASYNCcnt) with safe fallback.
// Probe result: builtin exists with prototype
//   void(v4i addrspace(1)* src, v4i addrspace(3)* dst, imm int off, imm int cpol)
// ---------------------------------------------------------------------------
#if defined(__has_builtin)
#if __has_builtin(__builtin_amdgcn_global_load_async_to_lds_b128)
#define TC_HAVE_ASYNC_LDS 1
#endif
#endif

__device__ __forceinline__ void stage_b128(_Float16* lds, const _Float16* g)
{
#if defined(TC_HAVE_ASYNC_LDS)
    __builtin_amdgcn_global_load_async_to_lds_b128(
        (__attribute__((address_space(1))) v4i*)g,
        (__attribute__((address_space(3))) v4i*)lds, 0, 0);
#else
    *(v8h*)lds = *(const v8h*)g;   // plain copy fallback (8 halves / 16B)
#endif
}

__device__ __forceinline__ void stage_wait()
{
#if defined(TC_HAVE_ASYNC_LDS)
#if __has_builtin(__builtin_amdgcn_s_wait_asynccnt)
    __builtin_amdgcn_s_wait_asynccnt(0);
#else
    asm volatile("s_wait_asynccnt 0x0" ::: "memory");
#endif
#endif
}

// ---------------------------------------------------------------------------
// Unified f16 WMMA GEMM:  C = epi( alpha * (A[M,K] * B[N,K]^T) + bias )
// A, B are f16 row-major with K contiguous. Double-buffered LDS, async
// staging, one barrier per k-step. Each wave: 32x32 output (4 WMMAs/step);
// block: 8 waves = 128(M) x 64(N) tile. Batched via grid.z = (zb, zh).
// Writes f32 C (if C32) and/or f16 C (if C16). EPI: 0 = bias, 1 = bias+relu.
// ---------------------------------------------------------------------------
template<int EPI>
__global__ __launch_bounds__(256)
void gemm_f16_wmma_kernel(const _Float16* __restrict__ A,  int lda, long long sAb, long long sAh,
                          const _Float16* __restrict__ Bm, int ldb, long long sBb, long long sBh,
                          const float* __restrict__ bias,
                          float*    __restrict__ C32, long long sCb,  long long sCh,
                          _Float16* __restrict__ C16, long long sC16b, long long sC16h,
                          int ldc, int M, int N, int K, int nh, float alpha)
{
    __shared__ __align__(16) _Float16 As[2][BM * LDP];
    __shared__ __align__(16) _Float16 Bs[2][BN * LDP];

    const int t    = threadIdx.x;
    const int lane = t & 31;
    const int wave = t >> 5;
    const int wm   = wave & 3;          // wave M index (0..3) -> 32 rows each
    const int wn   = wave >> 2;         // wave N index (0..1) -> 32 cols each
    const int lr   = lane & 15;
    const int hi   = (lane >> 4) & 1;

    const int m0 = blockIdx.y * BM;
    const int n0 = blockIdx.x * BN;
    const int z  = blockIdx.z;
    const int zb = z / nh;
    const int zh = z - zb * nh;

    const _Float16* Ab = A  + (long long)zb * sAb + (long long)zh * sAh;
    const _Float16* Bb = Bm + (long long)zb * sBb + (long long)zh * sBh;

    // staging coordinates (b128 = 8 halves per chunk)
    const int sar = t >> 1;             // A row 0..127, 2 threads/row
    const int sac = (t & 1) << 4;       // A col 0 or 16 (two chunks: +0, +8)
    const int sbr = t >> 2;             // B row 0..63, 4 threads/row
    const int sbc = (t & 3) << 3;       // B col 0,8,16,24
    int gbn = n0 + sbr; if (gbn > N - 1) gbn = N - 1;   // clamp (classifier N=10)

    v8f acc[2][2] = {};

    const int nk = K / BK;
    // prologue: stage k-block 0 into buffer 0
    {
        const _Float16* gA = Ab + (long long)(m0 + sar) * lda + sac;
        stage_b128(&As[0][sar * LDP + sac],     gA);
        stage_b128(&As[0][sar * LDP + sac + 8], gA + 8);
        const _Float16* gB = Bb + (long long)gbn * ldb + sbc;
        stage_b128(&Bs[0][sbr * LDP + sbc], gB);
    }

    for (int kb = 0; kb < nk; ++kb) {
        const int buf = kb & 1;
        stage_wait();
        __syncthreads();
        if (kb + 1 < nk) {
            const int k0 = (kb + 1) * BK;
            const _Float16* gA = Ab + (long long)(m0 + sar) * lda + (k0 + sac);
            stage_b128(&As[buf ^ 1][sar * LDP + sac],     gA);
            stage_b128(&As[buf ^ 1][sar * LDP + sac + 8], gA + 8);
            const _Float16* gB = Bb + (long long)gbn * ldb + (k0 + sbc);
            stage_b128(&Bs[buf ^ 1][sbr * LDP + sbc], gB);
        }

        // A fragments (16x32): lane<16 K{0-7,16-23}, lane>=16 K{8-15,24-31}
        v16h a[2];
        #pragma unroll
        for (int mi = 0; mi < 2; ++mi) {
            const int ro = (wm * 32 + mi * 16 + lr) * LDP + hi * 8;
            v8h a0 = *(const v8h*)&As[buf][ro];
            v8h a1 = *(const v8h*)&As[buf][ro + 16];
            #pragma unroll
            for (int i = 0; i < 8; ++i) { a[mi][i] = a0[i]; a[mi][i + 8] = a1[i]; }
        }
        // B fragments (32x16): lane<16 K{0-7,16-23}, lane>=16 K{8-15,24-31}
        #pragma unroll
        for (int ni = 0; ni < 2; ++ni) {
            const int ro = (wn * 32 + ni * 16 + lr) * LDP + hi * 8;
            v8h b0 = *(const v8h*)&Bs[buf][ro];
            v8h b1 = *(const v8h*)&Bs[buf][ro + 16];
            v16h b;
            #pragma unroll
            for (int i = 0; i < 8; ++i) { b[i] = b0[i]; b[i + 8] = b1[i]; }
            #pragma unroll
            for (int mi = 0; mi < 2; ++mi) {
                acc[mi][ni] = __builtin_amdgcn_wmma_f32_16x16x32_f16(
                    false, a[mi], false, b, (short)0, acc[mi][ni], false, false);
            }
        }
    }

    // epilogue: VGPR r holds M=r (lanes 0-15) / r+8 (lanes 16-31), N = lane&15
    float*    C32b = C32 ? C32 + (long long)zb * sCb  + (long long)zh * sCh  : nullptr;
    _Float16* C16b = C16 ? C16 + (long long)zb * sC16b + (long long)zh * sC16h : nullptr;
    const int gmb = m0 + wm * 32 + hi * 8;
    #pragma unroll
    for (int mi = 0; mi < 2; ++mi) {
        #pragma unroll
        for (int ni = 0; ni < 2; ++ni) {
            const int col = n0 + wn * 32 + ni * 16 + lr;
            if (col < N) {
                const float bv = bias ? bias[col] : 0.0f;
                #pragma unroll
                for (int r = 0; r < 8; ++r) {
                    float v = acc[mi][ni][r] * alpha + bv;
                    if (EPI == 1) v = fmaxf(v, 0.0f);
                    const long long row = gmb + mi * 16 + r;
                    if (C32b) C32b[row * ldc + col] = v;
                    if (C16b) C16b[row * ldc + col] = (_Float16)v;
                }
            }
        }
    }
}

// ---------------------------------------------------------------------------
// fp32 -> f16 conversion, 4 elements/thread (n must be a multiple of 1024)
// ---------------------------------------------------------------------------
__global__ __launch_bounds__(256)
void cvt_f32_f16_kernel(const float* __restrict__ in, _Float16* __restrict__ out)
{
    const int i = (blockIdx.x * 256 + threadIdx.x) << 2;
    float4 f = *(const float4*)(in + i);
    v4h h;
    h[0] = (_Float16)f.x; h[1] = (_Float16)f.y;
    h[2] = (_Float16)f.z; h[3] = (_Float16)f.w;
    *(v4h*)(out + i) = h;
}

// ---------------------------------------------------------------------------
// V transpose: vt16[(b*H+h)*HD + d][l] = qkv16[b,l, 2E + h*HD + d]
// ---------------------------------------------------------------------------
__global__ __launch_bounds__(256)
void vtrans_kernel(const _Float16* __restrict__ qkv16, _Float16* __restrict__ vt16)
{
    const int idx = blockIdx.x * 256 + threadIdx.x;      // 0 .. B*H*HD*L-1
    const int z   = idx >> 15;                            // / (HD*L)
    const int rem = idx & (TC_HD * TC_L - 1);
    const int d   = rem >> 9;                             // / L
    const int l   = rem & (TC_L - 1);
    const int b   = z >> 3;                               // / H
    const int h   = z & (TC_H - 1);
    vt16[idx] = qkv16[(size_t)(b * TC_L + l) * (3 * TC_E) + 2 * TC_E + h * TC_HD + d];
}

// ---------------------------------------------------------------------------
// Row softmax in place (f32) + f16 copy of the probabilities.
// ---------------------------------------------------------------------------
__global__ __launch_bounds__(256)
void softmax_kernel(float* __restrict__ S, _Float16* __restrict__ P16, int n)
{
    float*     row   = S   + (size_t)blockIdx.x * n;
    _Float16*  row16 = P16 + (size_t)blockIdx.x * n;
    __shared__ float red[256];
    const int tid = threadIdx.x;

    float m = -3.402823466e38f;
    for (int j = tid; j < n; j += 256) m = fmaxf(m, row[j]);
    red[tid] = m; __syncthreads();
    for (int s = 128; s > 0; s >>= 1) {
        if (tid < s) red[tid] = fmaxf(red[tid], red[tid + s]);
        __syncthreads();
    }
    m = red[0]; __syncthreads();

    float acc = 0.0f;
    for (int j = tid; j < n; j += 256) {
        float e = __expf(row[j] - m);
        row[j] = e;
        acc += e;
    }
    red[tid] = acc; __syncthreads();
    for (int s = 128; s > 0; s >>= 1) {
        if (tid < s) red[tid] += red[tid + s];
        __syncthreads();
    }
    const float inv = 1.0f / red[0];
    for (int j = tid; j < n; j += 256) {
        float v = row[j] * inv;
        row[j]   = v;
        row16[j] = (_Float16)v;
    }
}

// ---------------------------------------------------------------------------
// maps[b,i,j] = mean over heads of attn[b,h,i,j]
// ---------------------------------------------------------------------------
__global__ __launch_bounds__(256)
void head_mean_kernel(const float* __restrict__ S, float* __restrict__ outp)
{
    const int idx = blockIdx.x * 256 + threadIdx.x;      // 0 .. B*L*L-1
    const int b   = idx >> 18;
    const int rem = idx & ((TC_L * TC_L) - 1);
    const float* p = S + (size_t)b * TC_H * (TC_L * TC_L) + rem;
    float s = 0.0f;
    #pragma unroll
    for (int h = 0; h < TC_H; ++h) s += p[(size_t)h * (TC_L * TC_L)];
    outp[idx] = s * (1.0f / TC_H);
}

// ---------------------------------------------------------------------------
// h = LayerNorm(h + r) * g + b ; writes f32 h and f16 copy (E=512)
// ---------------------------------------------------------------------------
__global__ __launch_bounds__(256)
void add_ln_kernel(float* __restrict__ h, _Float16* __restrict__ h16,
                   const float* __restrict__ res,
                   const float* __restrict__ g, const float* __restrict__ bta)
{
    const int tid = threadIdx.x;
    const size_t base = (size_t)blockIdx.x * TC_E;
    float x0 = h[base + tid]       + res[base + tid];
    float x1 = h[base + tid + 256] + res[base + tid + 256];

    __shared__ float red[256];
    red[tid] = x0 + x1; __syncthreads();
    for (int s = 128; s > 0; s >>= 1) {
        if (tid < s) red[tid] += red[tid + s];
        __syncthreads();
    }
    const float mean = red[0] * (1.0f / TC_E);
    __syncthreads();

    red[tid] = x0 * x0 + x1 * x1; __syncthreads();
    for (int s = 128; s > 0; s >>= 1) {
        if (tid < s) red[tid] += red[tid + s];
        __syncthreads();
    }
    const float var = red[0] * (1.0f / TC_E) - mean * mean;
    const float inv = rsqrtf(var + 1e-5f);

    const float y0 = (x0 - mean) * inv * g[tid]       + bta[tid];
    const float y1 = (x1 - mean) * inv * g[tid + 256] + bta[tid + 256];
    h[base + tid]         = y0;
    h[base + tid + 256]   = y1;
    h16[base + tid]       = (_Float16)y0;
    h16[base + tid + 256] = (_Float16)y1;
}

// ---------------------------------------------------------------------------
template<int EPI>
static inline void launch_gemm(const _Float16* A,  int lda, long long sAb, long long sAh,
                               const _Float16* Bm, int ldb, long long sBb, long long sBh,
                               const float* bias,
                               float* C32, long long sCb, long long sCh,
                               _Float16* C16, long long sC16b, long long sC16h,
                               int ldc, int M, int N, int K, int nh, int nbatch,
                               float alpha, hipStream_t s)
{
    dim3 grid((N + BN - 1) / BN, M / BM, nbatch);
    gemm_f16_wmma_kernel<EPI><<<grid, 256, 0, s>>>(
        A, lda, sAb, sAh, Bm, ldb, sBb, sBh, bias,
        C32, sCb, sCh, C16, sC16b, sC16h, ldc, M, N, K, nh, alpha);
}

static inline void launch_cvt(const float* in, _Float16* out, int n, hipStream_t s)
{
    cvt_f32_f16_kernel<<<dim3(n / 1024), 256, 0, s>>>(in, out);
}

extern "C" void kernel_launch(void* const* d_in, const int* in_sizes, int n_in,
                              void* d_out, int out_size, void* d_ws, size_t ws_size,
                              hipStream_t stream)
{
    (void)in_sizes; (void)n_in; (void)out_size; (void)ws_size;

    const float* x    = (const float*)d_in[0];
    const float* w_in = (const float*)d_in[1];
    const float* b_in = (const float*)d_in[2];
    const float* wqkv = (const float*)d_in[3];
    const float* bqkv = (const float*)d_in[4];
    const float* wo   = (const float*)d_in[5];
    const float* bo   = (const float*)d_in[6];
    const float* ln1g = (const float*)d_in[7];
    const float* ln1b = (const float*)d_in[8];
    const float* w1   = (const float*)d_in[9];
    const float* b1   = (const float*)d_in[10];
    const float* w2   = (const float*)d_in[11];
    const float* b2   = (const float*)d_in[12];
    const float* ln2g = (const float*)d_in[13];
    const float* ln2b = (const float*)d_in[14];
    const float* wc   = (const float*)d_in[15];
    const float* bc   = (const float*)d_in[16];
    float* out = (float*)d_out;

    // ---- workspace layout ----
    const size_t M_ = 1024 * 1024;
    float* ws32 = (float*)d_ws;
    float* h32  = ws32;                 // 2M f32
    float* t32  = ws32 +  2 * M_;       // 2M f32
    float* sc32 = ws32 +  4 * M_;       // 16M f32   (ends at 20M f32 = 80MB)

    _Float16* f16p  = (_Float16*)(ws32 + 20 * M_);
    _Float16* x16   = f16p;  f16p += (size_t)TC_M * TC_DIN;        // 1M
    _Float16* h16   = f16p;  f16p += (size_t)TC_M * TC_E;          // 2M
    _Float16* qkv16 = f16p;  f16p += (size_t)TC_M * 3 * TC_E;      // 6M
    _Float16* att16 = f16p;  f16p += (size_t)TC_B * TC_H * TC_L * TC_L; // 16M
    _Float16* vt16  = f16p;  f16p += (size_t)TC_B * TC_H * TC_HD * TC_L; // 2M
    _Float16* o16   = f16p;  f16p += (size_t)TC_M * TC_E;          // 2M
    _Float16* ff16  = f16p;  f16p += (size_t)TC_M * TC_FF;         // 8M
    _Float16* win16 = f16p;  f16p += (size_t)TC_E * TC_DIN;
    _Float16* wqk16 = f16p;  f16p += (size_t)TC_NL * 3 * TC_E * TC_E;
    _Float16* wo16  = f16p;  f16p += (size_t)TC_NL * TC_E * TC_E;
    _Float16* w116  = f16p;  f16p += (size_t)TC_NL * TC_FF * TC_E;
    _Float16* w216  = f16p;  f16p += (size_t)TC_NL * TC_E * TC_FF;
    _Float16* wc16  = f16p;  f16p += (size_t)TC_C * TC_E;

    const long long LL     = (long long)TC_L * TC_L;
    const long long QKV_BS = (long long)TC_L * 3 * TC_E;
    const float     SCALE  = 0.125f;                         // 1/sqrt(64)
    const size_t    MAPS_OFF = (size_t)TC_B * TC_L * TC_C;   // 40960

    // ---- one-time fp32 -> f16 conversions (weights + input) ----
    launch_cvt(x,    x16,   TC_M * TC_DIN, stream);
    launch_cvt(w_in, win16, TC_E * TC_DIN, stream);
    launch_cvt(wqkv, wqk16, TC_NL * 3 * TC_E * TC_E, stream);
    launch_cvt(wo,   wo16,  TC_NL * TC_E * TC_E, stream);
    launch_cvt(w1,   w116,  TC_NL * TC_FF * TC_E, stream);
    launch_cvt(w2,   w216,  TC_NL * TC_E * TC_FF, stream);
    launch_cvt(wc,   wc16,  TC_C * TC_E, stream);

    // h = x @ w_in.T + b_in   (f32 residual stream + f16 copy)
    launch_gemm<0>(x16, TC_DIN, 0, 0, win16, TC_DIN, 0, 0, b_in,
                   h32, 0, 0, h16, 0, 0, TC_E,
                   TC_M, TC_E, TC_DIN, 1, 1, 1.0f, stream);

    for (int i = 0; i < TC_NL; ++i) {
        // qkv = h @ wqkv[i].T + bqkv[i]   (f16 only: feeds attention)
        launch_gemm<0>(h16, TC_E, 0, 0,
                       wqk16 + (size_t)i * 3 * TC_E * TC_E, TC_E, 0, 0,
                       bqkv + (size_t)i * 3 * TC_E,
                       nullptr, 0, 0, qkv16, 0, 0, 3 * TC_E,
                       TC_M, 3 * TC_E, TC_E, 1, 1, 1.0f, stream);

        // vt[z][d][l] = v  (makes P*V a row-major A*B^T GEMM)
        vtrans_kernel<<<dim3((TC_B * TC_H * TC_HD * TC_L) / 256), 256, 0, stream>>>(
            qkv16, vt16);

        // scores[b,h] = SCALE * (q @ k^T)   (f32 for softmax; alpha = SCALE)
        launch_gemm<0>(qkv16,        3 * TC_E, QKV_BS, TC_HD,
                       qkv16 + TC_E, 3 * TC_E, QKV_BS, TC_HD,
                       nullptr,
                       sc32, (long long)TC_H * LL, LL, nullptr, 0, 0, TC_L,
                       TC_L, TC_L, TC_HD, TC_H, TC_B * TC_H, SCALE, stream);

        // softmax rows (f32 in place) + f16 copy of probabilities
        softmax_kernel<<<dim3(TC_B * TC_H * TC_L), 256, 0, stream>>>(
            sc32, att16, TC_L);

        // maps[i] = mean over heads
        head_mean_kernel<<<dim3((TC_B * TC_L * TC_L) / 256), 256, 0, stream>>>(
            sc32, out + MAPS_OFF + (size_t)i * TC_B * TC_L * TC_L);

        // o[b,h] = attn @ v   == att16 @ vt16^T   (f16 only: feeds out-proj)
        launch_gemm<0>(att16, TC_L, (long long)TC_H * LL, LL,
                       vt16,  TC_L, (long long)TC_H * TC_HD * TC_L, (long long)TC_HD * TC_L,
                       nullptr,
                       nullptr, 0, 0,
                       o16, (long long)TC_L * TC_E, TC_HD, TC_E,
                       TC_L, TC_HD, TC_L, TC_H, TC_B * TC_H, 1.0f, stream);

        // t = o @ wo[i].T + bo[i]   (f32: feeds LN)
        launch_gemm<0>(o16, TC_E, 0, 0,
                       wo16 + (size_t)i * TC_E * TC_E, TC_E, 0, 0,
                       bo + (size_t)i * TC_E,
                       t32, 0, 0, nullptr, 0, 0, TC_E,
                       TC_M, TC_E, TC_E, 1, 1, 1.0f, stream);

        // h = LN(h + t)  -> f32 + f16
        add_ln_kernel<<<dim3(TC_M), 256, 0, stream>>>(
            h32, h16, t32, ln1g + (size_t)i * TC_E, ln1b + (size_t)i * TC_E);

        // ff = relu(h @ w1[i].T + b1[i])   (f16 only: feeds ffn2)
        launch_gemm<1>(h16, TC_E, 0, 0,
                       w116 + (size_t)i * TC_FF * TC_E, TC_E, 0, 0,
                       b1 + (size_t)i * TC_FF,
                       nullptr, 0, 0, ff16, 0, 0, TC_FF,
                       TC_M, TC_FF, TC_E, 1, 1, 1.0f, stream);

        // t = ff @ w2[i].T + b2[i]   (f32: feeds LN)
        launch_gemm<0>(ff16, TC_FF, 0, 0,
                       w216 + (size_t)i * TC_E * TC_FF, TC_FF, 0, 0,
                       b2 + (size_t)i * TC_E,
                       t32, 0, 0, nullptr, 0, 0, TC_E,
                       TC_M, TC_E, TC_FF, 1, 1, 1.0f, stream);

        // h = LN(h + t)  -> f32 + f16
        add_ln_kernel<<<dim3(TC_M), 256, 0, stream>>>(
            h32, h16, t32, ln2g + (size_t)i * TC_E, ln2b + (size_t)i * TC_E);
    }

    // logits = h @ wc.T + bc   (N=10: row clamp in staging, store guard)
    launch_gemm<0>(h16, TC_E, 0, 0, wc16, TC_E, 0, 0, bc,
                   out, 0, 0, nullptr, 0, 0, TC_C,
                   TC_M, TC_C, TC_E, 1, 1, 1.0f, stream);
}
